// GatedDeltaNet_14207751815226
// MI455X (gfx1250) — matrix operand order repaired and checked by reading, not compiled, not run on hardware
//
#include <hip/hip_runtime.h>
#include <hip/hip_bf16.h>

typedef __attribute__((ext_vector_type(16))) __bf16 v16bf;
typedef __attribute__((ext_vector_type(8)))  float  v8f;

// Problem constants (fixed by setup_inputs)
constexpr int Bb = 2, Ss = 1024, Hh = 2048;
constexpr int HV = 32, HK = 16, DK = 128, DV = 128, K4 = 4;
constexpr int KD = HK * DK;        // key_dim   = 2048
constexpr int VD = HV * DV;        // value_dim = 4096
constexpr int Cc = 2 * KD + VD;    // conv channels = 8192
constexpr float EPSN = 1e-6f;

// pack two f32 -> bf16x2 (lowers to v_cvt_pk_bf16_f32)
__device__ __forceinline__ unsigned int pkbf16(float a, float b) {
  unsigned short lo = __builtin_bit_cast(unsigned short, (__bf16)a);
  unsigned short hi = __builtin_bit_cast(unsigned short, (__bf16)b);
  return (unsigned int)lo | ((unsigned int)hi << 16);
}

// ---------------------------------------------------------------------------
// bf16 WMMA GEMM:  Out[n,m] = sum_k A[n,k] * W[m,k]
// mode 0: Out[n*M + m]            (row-major activations)
// mode 1: Out[(b*M + m)*S + s]    (channel-major, for conv input), n = b*S+s
// Block: 256 threads (8 waves). Tile: 128 rows (N) x 128 cols (M), K-step 32.
// Wave w owns rows [w*16, w*16+16) and iterates 8 col-tiles of 16.
// Staging: float4 global loads (coalesced) -> v_cvt_pk_bf16_f32 -> b32 LDS.
// ---------------------------------------------------------------------------
__global__ __launch_bounds__(256) void gemm_bf16_wmma(
    const float* __restrict__ A, const float* __restrict__ W,
    float* __restrict__ Out, int N, int M, int Kd, int mode, int S)
{
  __shared__ unsigned int ldsA[128 * 16];   // 128x32 bf16
  __shared__ unsigned int ldsB[128 * 16];   // 128x32 bf16
  const int tid  = threadIdx.x;
  const int lane = tid & 31;
  const int wv   = tid >> 5;
  const int nBase = blockIdx.y * 128;
  const int mBase = blockIdx.x * 128;

  v8f acc[8] = {};

  // Fragment addressing (contiguous k order k = 16*(lane/16) + s, same for A and B)
  const int ar   = lane & 15;               // A row within 16x32 tile
  const int acb  = (lane & 16) ? 16 : 0;    // A K-pair base
  const int bcol = lane & 15;               // B column
  const int kkb  = (lane & 16) ? 16 : 0;    // B K base

  // Staging assignment: 8 lanes sweep one 32-col row (128B), 32 rows/pass
  const int r0 = tid >> 3;                  // 0..31
  const int c4 = (tid & 7) << 2;            // 0,4,...,28

  for (int k0 = 0; k0 < Kd; k0 += 32) {
    #pragma unroll
    for (int i = 0; i < 4; ++i) {
      int row = r0 + 32 * i;
      const float4 va =
          *(const float4*)&A[(size_t)(nBase + row) * Kd + (k0 + c4)];
      int li = row * 16 + (c4 >> 1);
      ldsA[li]     = pkbf16(va.x, va.y);
      ldsA[li + 1] = pkbf16(va.z, va.w);
    }
    #pragma unroll
    for (int i = 0; i < 4; ++i) {
      int row = r0 + 32 * i;
      float4 vb = {0.f, 0.f, 0.f, 0.f};
      if ((mBase + row) < M)
        vb = *(const float4*)&W[(size_t)(mBase + row) * Kd + (k0 + c4)];
      int li = row * 16 + (c4 >> 1);
      ldsB[li]     = pkbf16(vb.x, vb.y);
      ldsB[li + 1] = pkbf16(vb.z, vb.w);
    }
    __syncthreads();

    // A fragment: two contiguous 8-element runs (merge to ds_load_b128)
    v16bf afrag;
    const __bf16* aRow = (const __bf16*)&ldsA[(wv * 16 + ar) * 16];
    #pragma unroll
    for (int e = 0; e < 8; ++e) afrag[e]     = aRow[acb + e];
    #pragma unroll
    for (int e = 0; e < 8; ++e) afrag[8 + e] = aRow[acb + 8 + e];

    #pragma unroll
    for (int nt = 0; nt < 8; ++nt) {
      v16bf bfrag;
      const __bf16* bRow =
          (const __bf16*)&ldsB[(nt * 16 + bcol) * 16] + kkb;
      #pragma unroll
      for (int e = 0; e < 16; ++e) bfrag[e] = bRow[e];
      acc[nt] = __builtin_amdgcn_wmma_f32_16x16x32_bf16(
          false, afrag, false, bfrag, (short)0, acc[nt], false, false);
    }
    __syncthreads();
  }

  // Epilogue: D layout col = lane&15, row = (lane<16 ? r : 8+r)
  #pragma unroll
  for (int nt = 0; nt < 8; ++nt) {
    int m = mBase + nt * 16 + (lane & 15);
    if (m < M) {
      #pragma unroll
      for (int r = 0; r < 8; ++r) {
        int row = nBase + wv * 16 + ((lane < 16) ? r : 8 + r);
        float val = acc[nt][r];
        if (mode == 0) {
          Out[(size_t)row * M + m] = val;
        } else {
          int b = row / S, s2 = row % S;
          Out[((size_t)b * M + m) * S + s2] = val;
        }
      }
    }
  }
}

// ---------------------------------------------------------------------------
// Depthwise causal conv (K=4) + SiLU; also emits new conv_state.
// qkv is [B, C, S] (channel-major from GEMM mode 1); output is [B, S, C].
// ---------------------------------------------------------------------------
__global__ __launch_bounds__(256) void conv_silu_kernel(
    const float* __restrict__ qkv_bcs, const float* __restrict__ conv_state,
    const float* __restrict__ conv_w, float* __restrict__ mixed_bsc,
    float* __restrict__ conv_state_out)
{
  int bc = blockIdx.x;                 // b*C + c
  int b = bc / Cc, c = bc % Cc;
  const float* x  = qkv_bcs + (size_t)bc * Ss;
  const float* cs = conv_state + (size_t)bc * (K4 - 1);
  float w[K4];
  #pragma unroll
  for (int k = 0; k < K4; ++k) w[k] = conv_w[c * K4 + k];

  for (int s = threadIdx.x; s < Ss; s += 256) {
    float acc = 0.f;
    #pragma unroll
    for (int k = 0; k < K4; ++k) {
      int t = s + k;                   // padded index
      float xv = (t < K4 - 1) ? cs[t] : x[t - (K4 - 1)];
      acc += xv * w[k];
    }
    float y = acc / (1.f + __expf(-acc));   // SiLU
    mixed_bsc[((size_t)b * Ss + s) * Cc + c] = y;
  }
  if (threadIdx.x < K4 - 1)
    conv_state_out[(size_t)bc * (K4 - 1) + threadIdx.x] =
        x[Ss - (K4 - 1) + threadIdx.x];
}

// ---------------------------------------------------------------------------
// Per-token prep: l2norm(q)*Dk^-0.5, l2norm(k), beta = sigmoid(b),
// g = -exp(A_log)*softplus(a + dt_bias). One block per (b,s).
// ---------------------------------------------------------------------------
__global__ __launch_bounds__(256) void prep_kernel(
    const float* __restrict__ mixed_bsc,
    const float* __restrict__ bbuf, const float* __restrict__ abuf,
    const float* __restrict__ dt_bias, const float* __restrict__ A_log,
    float* __restrict__ qn, float* __restrict__ kn,
    float* __restrict__ gArr, float* __restrict__ betaArr)
{
  __shared__ float redq[256], redk[256];
  int bs  = blockIdx.x;
  int tid = threadIdx.x;
  int d   = tid & 127, grp = tid >> 7;
  const float scale = rsqrtf((float)DK);
  const float* row = mixed_bsc + (size_t)bs * Cc;

  for (int it = 0; it < HK / 2; ++it) {
    int head = it * 2 + grp;
    float qv = row[head * DK + d];
    float kv = row[KD + head * DK + d];
    redq[tid] = qv * qv; redk[tid] = kv * kv;
    __syncthreads();
    for (int s = 64; s > 0; s >>= 1) {
      if ((tid & 127) < s) { redq[tid] += redq[tid + s]; redk[tid] += redk[tid + s]; }
      __syncthreads();
    }
    int base = grp << 7;
    qn[((size_t)bs * HK + head) * DK + d] = qv * rsqrtf(redq[base] + EPSN) * scale;
    kn[((size_t)bs * HK + head) * DK + d] = kv * rsqrtf(redk[base] + EPSN);
    __syncthreads();
  }

  if (tid < HV) {
    float bv = bbuf[(size_t)bs * HV + tid];
    float av = abuf[(size_t)bs * HV + tid] + dt_bias[tid];
    float sp = (av > 20.f) ? av : log1pf(__expf(av));
    gArr[(size_t)bs * HV + tid]    = -__expf(A_log[tid]) * sp;
    betaArr[(size_t)bs * HV + tid] = 1.f / (1.f + __expf(-bv));
  }
}

// ---------------------------------------------------------------------------
// Sequential delta-rule scan. One block per (b,h).
// State 128x128 f32 lives in REGISTERS: thread (v = tid&127, half = tid>>7)
// owns st[kk] = state[half*64+kk][v] (64 VGPRs). LDS carries only the shared
// per-step vectors (k_t, q_t, delta) read as broadcast float4, plus the 2-way
// column reduction. Inner loops are pure register FMA chains.
// ---------------------------------------------------------------------------
__global__ __launch_bounds__(256) void scan_kernel(
    const float* __restrict__ rec_state,
    const float* __restrict__ qn, const float* __restrict__ kn,
    const float* __restrict__ mixed_bsc,
    const float* __restrict__ gArr, const float* __restrict__ betaArr,
    float* __restrict__ core, float* __restrict__ final_state)
{
  __shared__ alignas(16) float kq[256];   // [0..127]=k_t, [128..255]=q_t
  __shared__ float red[256];
  __shared__ float dl[DV];

  int bh = blockIdx.x;
  int b = bh / HV, h = bh % HV;
  int hk = h / (HV / HK);
  int tid = threadIdx.x;
  int v = tid & 127, half = tid >> 7;
  int k0 = half * 64;

  // Register-resident state tile (coalesced loads: lanes sweep v)
  float st[64];
  #pragma unroll
  for (int kk = 0; kk < 64; ++kk)
    st[kk] = rec_state[(size_t)bh * DK * DV + (size_t)(k0 + kk) * DV + v];

  for (int t = 0; t < Ss; ++t) {
    size_t bsIdx = (size_t)b * Ss + t;
    if (half == 0) kq[v]       = kn[(bsIdx * HK + hk) * DK + v];
    else           kq[128 + v] = qn[(bsIdx * HK + hk) * DK + v];
    float eg   = __expf(gArr[bsIdx * HV + h]);
    float beta = betaArr[bsIdx * HV + h];
    __syncthreads();

    // pass A: decay state, accumulate kv_mem partial (k broadcast as float4)
    float accKV = 0.f;
    #pragma unroll
    for (int kk4 = 0; kk4 < 16; ++kk4) {
      const float4 kf = *(const float4*)&kq[k0 + kk4 * 4];
      st[kk4 * 4 + 0] *= eg; accKV += kf.x * st[kk4 * 4 + 0];
      st[kk4 * 4 + 1] *= eg; accKV += kf.y * st[kk4 * 4 + 1];
      st[kk4 * 4 + 2] *= eg; accKV += kf.z * st[kk4 * 4 + 2];
      st[kk4 * 4 + 3] *= eg; accKV += kf.w * st[kk4 * 4 + 3];
    }
    red[tid] = accKV;
    __syncthreads();

    if (half == 0) {
      float kvm = red[v] + red[128 + v];
      float vt  = mixed_bsc[bsIdx * Cc + 2 * KD + h * DV + v];
      dl[v] = (vt - kvm) * beta;
    }
    __syncthreads();

    // pass B: rank-1 update + output projection (k and q broadcast float4)
    float dmy = dl[v];
    float accO = 0.f;
    #pragma unroll
    for (int kk4 = 0; kk4 < 16; ++kk4) {
      const float4 kf = *(const float4*)&kq[k0 + kk4 * 4];
      const float4 qf = *(const float4*)&kq[128 + k0 + kk4 * 4];
      st[kk4 * 4 + 0] += kf.x * dmy; accO += qf.x * st[kk4 * 4 + 0];
      st[kk4 * 4 + 1] += kf.y * dmy; accO += qf.y * st[kk4 * 4 + 1];
      st[kk4 * 4 + 2] += kf.z * dmy; accO += qf.z * st[kk4 * 4 + 2];
      st[kk4 * 4 + 3] += kf.w * dmy; accO += qf.w * st[kk4 * 4 + 3];
    }
    red[tid] = accO;
    __syncthreads();

    if (half == 0)
      core[(bsIdx * HV + h) * DV + v] = red[v] + red[128 + v];
    __syncthreads();
  }

  #pragma unroll
  for (int kk = 0; kk < 64; ++kk)
    final_state[(size_t)bh * DK * DV + (size_t)(k0 + kk) * DV + v] = st[kk];
}

// ---------------------------------------------------------------------------
// RMS-norm over Dv + gate with SiLU(z). One block per (b,s,h), 128 threads.
// core, z, gated all share flat layout [(bs*HV+h)*DV + v].
// ---------------------------------------------------------------------------
__global__ __launch_bounds__(128) void normgate_kernel(
    const float* __restrict__ core, const float* __restrict__ z,
    const float* __restrict__ norm_weight, float* __restrict__ gated)
{
  __shared__ float red[128];
  int tid = threadIdx.x;
  size_t base = (size_t)blockIdx.x * DV;
  float val = core[base + tid];
  red[tid] = val * val;
  __syncthreads();
  for (int s = 64; s > 0; s >>= 1) {
    if (tid < s) red[tid] += red[tid + s];
    __syncthreads();
  }
  float var = red[0] * (1.f / DV);
  float xn  = val * rsqrtf(var + EPSN) * norm_weight[tid];
  float zv  = z[base + tid];
  gated[base + tid] = xn * (zv / (1.f + __expf(-zv)));
}

// ---------------------------------------------------------------------------
extern "C" void kernel_launch(void* const* d_in, const int* in_sizes, int n_in,
                              void* d_out, int out_size, void* d_ws, size_t ws_size,
                              hipStream_t stream) {
  const float* hidden     = (const float*)d_in[0];
  const float* conv_state = (const float*)d_in[1];
  const float* rec_state  = (const float*)d_in[2];
  const float* W_qkv      = (const float*)d_in[3];
  const float* W_z        = (const float*)d_in[4];
  const float* W_b        = (const float*)d_in[5];
  const float* W_a        = (const float*)d_in[6];
  const float* conv_w     = (const float*)d_in[7];
  const float* dt_bias    = (const float*)d_in[8];
  const float* A_log      = (const float*)d_in[9];
  const float* norm_w     = (const float*)d_in[10];
  const float* W_out      = (const float*)d_in[11];

  float* out        = (float*)d_out;
  float* out_main   = out;                                   // B*S*H
  float* out_conv   = out_main + (size_t)Bb * Ss * Hh;       // B*C*(K-1)
  float* out_final  = out_conv + (size_t)Bb * Cc * (K4 - 1); // B*Hv*Dk*Dv

  float* ws = (float*)d_ws;
  size_t oQKV  = 0;
  size_t oZ    = oQKV  + (size_t)Bb * Cc * Ss;
  size_t oBv   = oZ    + (size_t)Bb * Ss * VD;
  size_t oAv   = oBv   + (size_t)Bb * Ss * HV;
  size_t oMix  = oAv   + (size_t)Bb * Ss * HV;
  size_t oQN   = oMix  + (size_t)Bb * Ss * Cc;
  size_t oKN   = oQN   + (size_t)Bb * Ss * HK * DK;
  size_t oG    = oKN   + (size_t)Bb * Ss * HK * DK;
  size_t oBeta = oG    + (size_t)Bb * Ss * HV;
  size_t oCore = oBeta + (size_t)Bb * Ss * HV;
  float* gated = ws + oQKV;   // alias: qkv buffer dead after conv kernel

  const int N = Bb * Ss;      // 2048 token rows
  dim3 blk(256);

  // 1-4: projections (bf16 WMMA, 128x128 block tiles)
  gemm_bf16_wmma<<<dim3(Cc / 128, N / 128), blk, 0, stream>>>(
      hidden, W_qkv, ws + oQKV, N, Cc, Hh, 1, Ss);
  gemm_bf16_wmma<<<dim3(VD / 128, N / 128), blk, 0, stream>>>(
      hidden, W_z, ws + oZ, N, VD, Hh, 0, Ss);
  gemm_bf16_wmma<<<dim3(1, N / 128), blk, 0, stream>>>(
      hidden, W_b, ws + oBv, N, HV, Hh, 0, Ss);
  gemm_bf16_wmma<<<dim3(1, N / 128), blk, 0, stream>>>(
      hidden, W_a, ws + oAv, N, HV, Hh, 0, Ss);

  // 5: depthwise conv + SiLU (+ new conv_state)
  conv_silu_kernel<<<Bb * Cc, 256, 0, stream>>>(
      ws + oQKV, conv_state, conv_w, ws + oMix, out_conv);

  // 6: l2norm q/k, beta, g
  prep_kernel<<<N, 256, 0, stream>>>(
      ws + oMix, ws + oBv, ws + oAv, dt_bias, A_log,
      ws + oQN, ws + oKN, ws + oG, ws + oBeta);

  // 7: sequential delta-rule scan (register-resident state)
  scan_kernel<<<Bb * HV, 256, 0, stream>>>(
      rec_state, ws + oQN, ws + oKN, ws + oMix, ws + oG, ws + oBeta,
      ws + oCore, out_final);

  // 8: RMS-norm + SiLU gate
  normgate_kernel<<<N * HV, 128, 0, stream>>>(
      ws + oCore, ws + oZ, norm_w, gated);

  // 9: output projection (bf16 WMMA)
  gemm_bf16_wmma<<<dim3(Hh / 128, N / 128), blk, 0, stream>>>(
      gated, W_out, out_main, N, Hh, VD, 0, Ss);
}
